// MultiHeadSelfAttention_1803886265532
// MI455X (gfx1250) — matrix operand order repaired and checked by reading, not compiled, not run on hardware
//
#include <hip/hip_runtime.h>
#include <hip/hip_bf16.h>
#include <stdint.h>

typedef __bf16 bf16_t;
typedef __attribute__((ext_vector_type(16))) __bf16 v16bf;
typedef __attribute__((ext_vector_type(8)))  float  v8f;
typedef __attribute__((ext_vector_type(4)))  unsigned int v4u;
typedef __attribute__((ext_vector_type(8)))  int v8i;
typedef __attribute__((ext_vector_type(4)))  int v4i;

#define BDIM 8
#define CDIM 1024
#define EDIM 768
#define HDIM 12

// ---- CDNA5 async Global->LDS copy (16 bytes per lane), tracked by ASYNCcnt ----
static __device__ __forceinline__ void async_copy16(void* lds, const void* g) {
    uint32_t l = (uint32_t)(uintptr_t)lds;          // low 32 bits = LDS byte offset
    asm volatile("global_load_async_to_lds_b128 %0, %1, off"
                 :: "v"(l), "v"((uint64_t)(uintptr_t)g)
                 : "memory");
}
#define ASYNC_WAIT_0 asm volatile("s_wait_asynccnt 0x0" ::: "memory")

// ---- CDNA5 Tensor Data Mover: 2D tile DMA Global->LDS, tracked by TENSORcnt ----
// rows x rowlen elements of 2 bytes; global row stride = stride_elems elements.
// Destination LDS is packed [rows][rowlen].
static __device__ __forceinline__ void tdm_load_2d(void* lds, const void* g,
                                                   uint32_t rowlen, uint32_t rows,
                                                   uint32_t stride_elems) {
    uint64_t ga = (uint64_t)(uintptr_t)g;
    v4u g0 = { 1u,                                   // count=1 (valid descriptor)
               (uint32_t)(uintptr_t)lds,             // lds_addr (bytes)
               (uint32_t)ga,                         // global_addr[31:0]
               (uint32_t)((ga >> 32) & 0x1FFFFFFu) | 0x80000000u }; // addr[56:32] | type=2
    v8i g1;
    g1[0] = (int)(1u << 16);            // workgroup_mask=0, data_size=1 (2 bytes)
    g1[1] = (int)(rowlen << 16);        // tensor_dim0[15:0]   (bits 79:48)
    g1[2] = (int)(rows << 16);          // tensor_dim0 hi=0 | tensor_dim1[15:0] (111:80)
    g1[3] = (int)(rowlen << 16);        // tensor_dim1 hi=0 | tile_dim0 (127:112)
    g1[4] = (int)rows;                  // tile_dim1 (143:128), tile_dim2=0
    g1[5] = (int)stride_elems;          // tensor_dim0_stride[31:0] (207:160)
    g1[6] = 0;                          // stride hi=0, tensor_dim1_stride lo=0
    g1[7] = 0;
    v4i zz = {0, 0, 0, 0};
    v8i z8i = {0, 0, 0, 0, 0, 0, 0, 0};
    __builtin_amdgcn_tensor_load_to_lds(g0, g1, zz, zz, z8i, 0);
}
#define TENSOR_WAIT_0 __builtin_amdgcn_s_wait_tensorcnt((short)0)
#define TENSOR_WAIT_2 __builtin_amdgcn_s_wait_tensorcnt((short)2)

// Build a 16x32 (A) or 32x16 (B) bf16 WMMA fragment from two contiguous
// 8-element (16-byte) runs, lane half h getting k = 16h + s (contiguous order, same for A and B).
static __device__ __forceinline__ v16bf frag2(const bf16_t* p0, const bf16_t* p1) {
    union { uint4 u[2]; v16bf v; } f;
    f.u[0] = *(const uint4*)p0;
    f.u[1] = *(const uint4*)p1;
    return f.v;
}

__global__ void cvt_f32_bf16(const float* __restrict__ in, bf16_t* __restrict__ out, int n4) {
    int i = blockIdx.x * blockDim.x + threadIdx.x;
    if (i < n4) {
        float4 f = ((const float4*)in)[i];
        union { bf16_t h[4]; uint2 u; } o;
        o.h[0] = (bf16_t)f.x; o.h[1] = (bf16_t)f.y;
        o.h[2] = (bf16_t)f.z; o.h[3] = (bf16_t)f.w;
        ((uint2*)out)[i] = o.u;
    }
}

// Transpose-convert weights: in = f32 [K][N] row-major, out = bf16 [N][K] row-major.
__global__ void cvt_transpose_bf16(const float* __restrict__ in, bf16_t* __restrict__ out,
                                   int K, int N, int total /* = N*K/8 */) {
    int idx = blockIdx.x * blockDim.x + threadIdx.x;
    if (idx >= total) return;
    int kg8 = K >> 3;
    int n  = idx / kg8;
    int kg = (idx - n * kg8) * 8;
    union { bf16_t h[8]; uint4 u; } o;
#pragma unroll
    for (int e = 0; e < 8; e++)
        o.h[e] = (bf16_t)in[(size_t)(kg + e) * N + n];
    *(uint4*)&out[(size_t)n * K + kg] = o.u;
}

// Generic bf16 GEMM: out = A[MxK] * Wt^T + bias  (Wt is bf16 [N][K] row-major).
// Tile staging via TDM (tensor_load_to_lds), double-buffered:
// wave 0 issues 2 tile DMAs per stage; s_wait_tensorcnt 2 overlaps stage k+1
// DMA with stage k WMMAs.
// MODE 0: out = float [M x N]                      (final projection -> d_out)
// MODE 1: out = bf16  [B,H,C,64]   (q/k heads)     col = h*64 + qs, row = b*C + c
// MODE 2: out = bf16  [B,H,768,C]  (v, TRANSPOSED) col = h*768 + e, row = b*C + c
template <int MODE>
__global__ __launch_bounds__(256)
void gemm_bf16_wmma(const bf16_t* __restrict__ A, const bf16_t* __restrict__ Wt,
                    const float* __restrict__ bias, void* __restrict__ outp,
                    int M, int N, int K) {
    __shared__ __align__(16) bf16_t As[2][128][32];   // [buf][m][k]  8KB x2
    __shared__ __align__(16) bf16_t Bs[2][128][32];   // [buf][n][k]  8KB x2
    const int tid  = threadIdx.x;
    const int lane = tid & 31;
    const int wave = tid >> 5;
    const int wm = wave >> 2;            // 0..1  (64 rows each)
    const int wn = wave & 3;             // 0..3  (32 cols each)
    const int m0 = blockIdx.y * 128;
    const int n0 = blockIdx.x * 128;

    const v8f z8 = {0.f,0.f,0.f,0.f,0.f,0.f,0.f,0.f};
    v8f acc[4][2];
#pragma unroll
    for (int i = 0; i < 4; i++)
#pragma unroll
        for (int j = 0; j < 2; j++) acc[i][j] = z8;

    const int koff = (lane < 16) ? 0 : 16;   // A-frag per-lane k offset
    const int kb   = (lane < 16) ? 0 : 16;   // B-frag per-lane k offset

    // Stage one 128x32 A tile + 128x32 B tile into buffer `buf` (TDM, wave 0 only).
    auto stage = [&](int buf, int k0) {
        tdm_load_2d(&As[buf][0][0], &A[(size_t)m0 * K + k0], 32, 128, (uint32_t)K);
        tdm_load_2d(&Bs[buf][0][0], &Wt[(size_t)n0 * K + k0], 32, 128, (uint32_t)K);
    };

    if (wave == 0) stage(0, 0);
    int buf = 0;
    for (int k0 = 0; k0 < K; k0 += 32, buf ^= 1) {
        bool has_next = (k0 + 32) < K;
        if (wave == 0) {
            if (has_next) { stage(buf ^ 1, k0 + 32); TENSOR_WAIT_2; }
            else          { TENSOR_WAIT_0; }
        }
        __syncthreads();

        v16bf af[4], bf[2];
#pragma unroll
        for (int mt = 0; mt < 4; mt++) {
            const bf16_t* p = &As[buf][wm * 64 + mt * 16 + (lane & 15)][0];
            af[mt] = frag2(p + koff, p + koff + 8);
        }
#pragma unroll
        for (int nt = 0; nt < 2; nt++) {
            const bf16_t* p = &Bs[buf][wn * 32 + nt * 16 + (lane & 15)][0];
            bf[nt] = frag2(p + kb, p + kb + 8);
        }
#pragma unroll
        for (int mt = 0; mt < 4; mt++)
#pragma unroll
            for (int nt = 0; nt < 2; nt++)
                acc[mt][nt] = __builtin_amdgcn_wmma_f32_16x16x32_bf16(
                    false, af[mt], false, bf[nt], (short)0, acc[mt][nt], false, false);
        __syncthreads();   // protect buffer being refilled next iteration
    }

    // Epilogue: C/D layout => lane = column, VGPR r = row (+8 for lanes 16..31)
#pragma unroll
    for (int nt = 0; nt < 2; nt++) {
        int col = n0 + wn * 32 + nt * 16 + (lane & 15);
        float bv = bias[col];
#pragma unroll
        for (int mt = 0; mt < 4; mt++) {
#pragma unroll
            for (int r = 0; r < 8; r++) {
                int row = m0 + wm * 64 + mt * 16 + r + ((lane < 16) ? 0 : 8);
                float val = acc[mt][nt][r] + bv;
                if (MODE == 0) {
                    ((float*)outp)[(size_t)row * N + col] = val;
                } else if (MODE == 1) {
                    int b = row >> 10, c = row & 1023;
                    int h = col >> 6,  qs = col & 63;
                    ((bf16_t*)outp)[((((size_t)b * HDIM + h) * CDIM + c) << 6) + qs] = (bf16_t)val;
                } else {
                    int b = row >> 10, c = row & 1023;
                    int h = col / EDIM, e = col % EDIM;
                    ((bf16_t*)outp)[(((size_t)b * HDIM + h) * EDIM + e) * CDIM + c] = (bf16_t)val;
                }
            }
        }
    }
}

// Flash-style causal attention (two-pass: max, then exp/sum + P*V).
// Grid: x = B*H (96), y = C/64 (16). 8 waves: qt = wave&3 (16 q-rows each),
// eh = wave>>2 selects an E half (384 cols = 24 WMMA n-tiles).
// q,k: [B,H,C,64] bf16; v: [B,H,768,C] bf16 (transposed); out: [B,C,H*768] bf16.
// K tile staged with per-lane async copies (ASYNCcnt); V tile (48KB, strided 2D)
// staged with one TDM DMA per step (TENSORcnt).
__global__ __launch_bounds__(256)
void attn_flash(const bf16_t* __restrict__ qg, const bf16_t* __restrict__ kg,
                const bf16_t* __restrict__ vg, bf16_t* __restrict__ outf) {
    __shared__ __align__(16) bf16_t Kt[32][64];      // 4 KB   [key][d]
    __shared__ __align__(16) bf16_t Vt[768][32];     // 48 KB  [e][key]
    __shared__ __align__(16) bf16_t Pt[4][16][32];   // 4 KB   [qt][row][key]
    __shared__ float Ls[64];

    const int tid = threadIdx.x, lane = tid & 31, wave = tid >> 5;
    const int qt = wave & 3, eh = wave >> 2;
    const int bh = blockIdx.x;
    const int b = bh / HDIM, h = bh % HDIM;
    const int qblk = blockIdx.y;
    const int q0 = qblk * 64 + qt * 16;
    const float scale = 0.125f;  // 1/sqrt(64)

    const bf16_t* qb = qg + ((size_t)bh * CDIM) * 64;
    const bf16_t* kp = kg + ((size_t)bh * CDIM) * 64;
    const bf16_t* vb = vg + (size_t)bh * EDIM * CDIM;

    const int koff = (lane < 16) ? 0 : 16;
    const int kb2  = (lane < 16) ? 0 : 16;
    const v8f z8 = {0.f,0.f,0.f,0.f,0.f,0.f,0.f,0.f};

    // Resident Q fragments (d = 0..31 and 32..63)
    const bf16_t* qrow = qb + (size_t)(q0 + (lane & 15)) * 64;
    v16bf qf0 = frag2(qrow + 0  + koff, qrow + 0  + koff + 8);
    v16bf qf1 = frag2(qrow + 32 + koff, qrow + 32 + koff + 8);

    const int jmax = qblk * 2 + 1;   // causal: key tiles of 32 up to query block

    float mrow[8];
#pragma unroll
    for (int r = 0; r < 8; r++) mrow[r] = -1e30f;

    // ---------------- Pass A: row maxima ----------------
    for (int j = 0; j <= jmax; j++) {
        {   // K tile 32x64 = 256 b128 copies, one per thread (async)
            int kk = tid >> 3, g = tid & 7;
            async_copy16(&Kt[kk][g * 8], &kp[(size_t)(j * 32 + kk) * 64 + g * 8]);
        }
        ASYNC_WAIT_0;
        __syncthreads();
        if (eh == 0) {
#pragma unroll
            for (int nt = 0; nt < 2; nt++) {
                const bf16_t* kr = &Kt[nt * 16 + (lane & 15)][0];
                v16bf b0 = frag2(kr + 0  + kb2, kr + 0  + kb2 + 8);
                v16bf b1 = frag2(kr + 32 + kb2, kr + 32 + kb2 + 8);
                v8f s = z8;
                s = __builtin_amdgcn_wmma_f32_16x16x32_bf16(false, qf0, false, b0, (short)0, s, false, false);
                s = __builtin_amdgcn_wmma_f32_16x16x32_bf16(false, qf1, false, b1, (short)0, s, false, false);
#pragma unroll
                for (int r = 0; r < 8; r++) {
                    int rowg = q0 + r + ((lane < 16) ? 0 : 8);
                    int colg = j * 32 + nt * 16 + (lane & 15);
                    float v = s[r] * scale;
                    if (colg > rowg) v = -1e30f;
                    mrow[r] = fmaxf(mrow[r], v);
                }
            }
        }
        __syncthreads();
    }
    if (eh == 0) {
#pragma unroll
        for (int r = 0; r < 8; r++) {
            float m = mrow[r];
            m = fmaxf(m, __shfl_xor(m, 1, 32));
            m = fmaxf(m, __shfl_xor(m, 2, 32));
            m = fmaxf(m, __shfl_xor(m, 4, 32));
            m = fmaxf(m, __shfl_xor(m, 8, 32));
            mrow[r] = m;
        }
    }

    // ---------------- Pass B: P = exp(S - m), out += P*V ----------------
    v8f acc[24];
#pragma unroll
    for (int i = 0; i < 24; i++) acc[i] = z8;
    float lrow[8];
#pragma unroll
    for (int r = 0; r < 8; r++) lrow[r] = 0.f;

    for (int j = 0; j <= jmax; j++) {
        // V tile 768x32 via TDM: rows e=0..767, row stride CDIM elements
        if (wave == 0)
            tdm_load_2d(&Vt[0][0], vb + (size_t)j * 32, 32, 768, CDIM);
        {   // K tile (per-lane async)
            int kk = tid >> 3, g = tid & 7;
            async_copy16(&Kt[kk][g * 8], &kp[(size_t)(j * 32 + kk) * 64 + g * 8]);
        }
        ASYNC_WAIT_0;
        if (wave == 0) TENSOR_WAIT_0;
        __syncthreads();

        if (eh == 0) {
#pragma unroll
            for (int nt = 0; nt < 2; nt++) {
                const bf16_t* kr = &Kt[nt * 16 + (lane & 15)][0];
                v16bf b0 = frag2(kr + 0  + kb2, kr + 0  + kb2 + 8);
                v16bf b1 = frag2(kr + 32 + kb2, kr + 32 + kb2 + 8);
                v8f s = z8;
                s = __builtin_amdgcn_wmma_f32_16x16x32_bf16(false, qf0, false, b0, (short)0, s, false, false);
                s = __builtin_amdgcn_wmma_f32_16x16x32_bf16(false, qf1, false, b1, (short)0, s, false, false);
#pragma unroll
                for (int r = 0; r < 8; r++) {
                    int prow = r + ((lane < 16) ? 0 : 8);
                    int rowg = q0 + prow;
                    int colg = j * 32 + nt * 16 + (lane & 15);
                    float v = s[r] * scale;
                    if (colg > rowg) v = -1e30f;
                    float p = __expf(v - mrow[r]);
                    lrow[r] += p;
                    Pt[qt][prow][nt * 16 + (lane & 15)] = (bf16_t)p;
                }
            }
        }
        __syncthreads();

        // P (16x32) * V (32x384-half): 24 WMMAs per wave
        const bf16_t* pp = &Pt[qt][lane & 15][0];
        v16bf pf = frag2(pp + koff, pp + koff + 8);
#pragma unroll
        for (int nt2 = 0; nt2 < 24; nt2++) {
            const bf16_t* vp = &Vt[eh * 384 + nt2 * 16 + (lane & 15)][0];
            v16bf bv = frag2(vp + kb2, vp + kb2 + 8);
            acc[nt2] = __builtin_amdgcn_wmma_f32_16x16x32_bf16(
                false, pf, false, bv, (short)0, acc[nt2], false, false);
        }
        __syncthreads();
    }

    // row sums -> LDS, then normalize + store
    if (eh == 0) {
#pragma unroll
        for (int r = 0; r < 8; r++) {
            float l = lrow[r];
            l += __shfl_xor(l, 1, 32);
            l += __shfl_xor(l, 2, 32);
            l += __shfl_xor(l, 4, 32);
            l += __shfl_xor(l, 8, 32);
            if ((lane & 15) == 0)
                Ls[qt * 16 + r + ((lane < 16) ? 0 : 8)] = l;
        }
    }
    __syncthreads();

#pragma unroll
    for (int nt2 = 0; nt2 < 24; nt2++) {
        int e = h * EDIM + eh * 384 + nt2 * 16 + (lane & 15);
#pragma unroll
        for (int r = 0; r < 8; r++) {
            int row = r + ((lane < 16) ? 0 : 8);
            float linv = 1.0f / Ls[qt * 16 + row];
            size_t idx = (size_t)(b * CDIM + q0 + row) * (HDIM * EDIM) + e;
            outf[idx] = (bf16_t)(acc[nt2][r] * linv);
        }
    }
}

extern "C" void kernel_launch(void* const* d_in, const int* in_sizes, int n_in,
                              void* d_out, int out_size, void* d_ws, size_t ws_size,
                              hipStream_t stream) {
    (void)in_sizes; (void)n_in; (void)out_size; (void)ws_size;
    const float* x  = (const float*)d_in[0];
    const float* Wk = (const float*)d_in[1];
    const float* bk = (const float*)d_in[2];
    const float* Wq = (const float*)d_in[3];
    const float* bq = (const float*)d_in[4];
    const float* Wv = (const float*)d_in[5];
    const float* bv = (const float*)d_in[6];
    const float* Wo = (const float*)d_in[7];
    const float* bo = (const float*)d_in[8];
    float* out = (float*)d_out;

    char* ws = (char*)d_ws;
    size_t off = 0;
    auto alloc = [&](size_t bytes) {
        char* p = ws + off;
        off += (bytes + 255) & ~(size_t)255;
        return p;
    };
    const size_t M = (size_t)BDIM * CDIM;  // 8192
    bf16_t* xb   = (bf16_t*)alloc(M * EDIM * 2);
    bf16_t* Wkb  = (bf16_t*)alloc((size_t)EDIM * 768 * 2);   // [N][K] transposed
    bf16_t* Wqb  = (bf16_t*)alloc((size_t)EDIM * 768 * 2);
    bf16_t* Wvb  = (bf16_t*)alloc((size_t)EDIM * 9216 * 2);
    bf16_t* Wob  = (bf16_t*)alloc((size_t)9216 * EDIM * 2);
    bf16_t* qbuf = (bf16_t*)alloc((size_t)BDIM * HDIM * CDIM * 64 * 2);
    bf16_t* kbuf = (bf16_t*)alloc((size_t)BDIM * HDIM * CDIM * 64 * 2);
    bf16_t* vbuf = (bf16_t*)alloc((size_t)BDIM * HDIM * EDIM * CDIM * 2);
    bf16_t* aflt = (bf16_t*)alloc(M * 9216 * 2);

    {   // x: plain convert (stays [M][K] row-major)
        int n4 = (int)((M * EDIM) / 4);
        cvt_f32_bf16<<<(n4 + 255) / 256, 256, 0, stream>>>(x, xb, n4);
    }
    auto cvtT = [&](const float* s, bf16_t* d, int K, int N) {
        int total = (int)(((size_t)N * K) / 8);
        cvt_transpose_bf16<<<(total + 255) / 256, 256, 0, stream>>>(s, d, K, N, total);
    };
    cvtT(Wk, Wkb, 768, 768);     // Wk f32 [768][768]  -> bf16 [768][768]^T
    cvtT(Wq, Wqb, 768, 768);
    cvtT(Wv, Wvb, 768, 9216);    // -> bf16 [9216][768]
    cvtT(Wo, Wob, 9216, 768);    // -> bf16 [768][9216]

    dim3 blk(256);
    // Projections (k, q, v) — v written transposed per (b,h)
    gemm_bf16_wmma<1><<<dim3(768 / 128,  (int)M / 128), blk, 0, stream>>>(xb, Wkb, bk, kbuf, (int)M, 768,  768);
    gemm_bf16_wmma<1><<<dim3(768 / 128,  (int)M / 128), blk, 0, stream>>>(xb, Wqb, bq, qbuf, (int)M, 768,  768);
    gemm_bf16_wmma<2><<<dim3(9216 / 128, (int)M / 128), blk, 0, stream>>>(xb, Wvb, bv, vbuf, (int)M, 9216, 768);
    // Attention
    attn_flash<<<dim3(BDIM * HDIM, CDIM / 64), blk, 0, stream>>>(qbuf, kbuf, vbuf, aflt);
    // Output projection -> f32 d_out
    gemm_bf16_wmma<0><<<dim3(768 / 128, (int)M / 128), blk, 0, stream>>>(aflt, Wob, bo, out, (int)M, 768, 9216);
}
